// CausalGNN_16569983828591
// MI455X (gfx1250) — matrix operand order repaired and checked by reading, not compiled, not run on hardware
//
#include <hip/hip_runtime.h>
#include <hip/hip_bf16.h>
#include <math.h>

// B=8, T=512, N=64, D=64; tokens = B*T = 4096
typedef __attribute__((ext_vector_type(16))) __bf16 v16bf;
typedef __attribute__((ext_vector_type(8)))  __bf16 v8bf;
typedef __attribute__((ext_vector_type(8)))  float  v8f;

#define LDS_P 72  // padded pitch (bf16 elems): 144B rows, 16B aligned, bank-skewed

// Branch-free fast tanh: tanh(x) = 1 - 2/(exp(2x)+1).
// exp(2x)->inf  => 1 ; exp(2x)->0 => -1. v_exp_f32 + v_rcp_f32, no divergence.
__device__ __forceinline__ float tanh_fast(float x) {
  float e = __expf(2.0f * x);
  return 1.0f - 2.0f * __builtin_amdgcn_rcpf(e + 1.0f);
}

__device__ __forceinline__ v16bf cat8(v8bf lo, v8bf hi) {
  return __builtin_shufflevector(lo, hi, 0,1,2,3,4,5,6,7,8,9,10,11,12,13,14,15);
}

// A fragment (16x32 bf16): rows m (per-lane), cols kt*32..+31, from row-major LDS [m][k].
// Layout per ISA: lanes 0-15 take K {0..7,16..23}, lanes 16-31 take K {8..15,24..31}.
__device__ __forceinline__ v16bf load_a_frag(const __bf16* arr, int m, int kt, int hl) {
  const __bf16* p = arr + m * LDS_P + kt * 32 + hl * 8;
  v8bf lo = *(const v8bf*)(p);
  v8bf hi = *(const v8bf*)(p + 16);
  return cat8(lo, hi);
}

// B fragment (32x16 bf16) of B[k][n], stored TRANSPOSED in LDS as bt[n][k]:
// lane's column n = nt*16+lr, same K order as A: hl=0 -> K {0..7,16..23}, hl=1 -> {8..15,24..31}.
__device__ __forceinline__ v16bf load_b_frag(const __bf16* bt, int nt, int kt, int lr, int hl) {
  const __bf16* p = bt + (nt * 16 + lr) * LDS_P + kt * 32 + hl * 8;
  v8bf lo = *(const v8bf*)(p);
  v8bf hi = *(const v8bf*)(p + 16);
  return cat8(lo, hi);
}

#define WMMA_BF16(acc, a, b) \
  __builtin_amdgcn_wmma_f32_16x16x32_bf16(false, (a), false, (b), (short)0, (acc), false, false)

// ---- prep: fold sigmoid + self-loop mask + 1/parent_count into bf16 Ahat;
//      split upd_w into its two 64x64 halves; convert msg_w. Stored row-major [e][d]
//      (== B^T for the GEMMs, which is exactly what load_b_frag wants).
__global__ void __launch_bounds__(256) gnn_prep_kernel(
    const float* __restrict__ W, const float* __restrict__ msg_w,
    const float* __restrict__ upd_w, __bf16* __restrict__ wsb)
{
  int tid = threadIdx.x;
  if (tid < 64) {
    int i = tid;
    float s = 0.f;
    for (int j = 0; j < 64; ++j) {
      float a = (j == i) ? 0.f : __builtin_amdgcn_rcpf(1.f + __expf(-W[i * 64 + j]));
      s += a;
    }
    float inv = 1.f / fmaxf(s, 1e-8f);
    for (int j = 0; j < 64; ++j) {
      float a = (j == i) ? 0.f : __builtin_amdgcn_rcpf(1.f + __expf(-W[i * 64 + j]));
      wsb[i * 64 + j] = (__bf16)(a * inv);
    }
  }
  for (int f = tid; f < 4096; f += 256) {
    int e = f >> 6, d = f & 63;
    wsb[4096 + f]     = (__bf16)msg_w[f];                  // msg_w[e][d]
    wsb[2 * 4096 + f] = (__bf16)upd_w[e * 128 + d];        // upd_w[:, :64]
    wsb[3 * 4096 + f] = (__bf16)upd_w[e * 128 + 64 + d];   // upd_w[:, 64:]
  }
}

// ---- main fused kernel: one 4-wave WG per token; out kept in registers (C layout).
__global__ void __launch_bounds__(128) gnn_main_kernel(
    const float* __restrict__ latent, const float* __restrict__ msg_b,
    const float* __restrict__ upd_b, const __bf16* __restrict__ wsb,
    float* __restrict__ out)
{
  __shared__ __attribute__((aligned(16))) __bf16 sAb[64 * LDS_P];    // Ahat, A-operand [i][j]
  __shared__ __attribute__((aligned(16))) __bf16 sWmsg[64 * LDS_P];  // msg_w [e][d] (B^T)
  __shared__ __attribute__((aligned(16))) __bf16 sWu1[64 * LDS_P];   // upd_w[:, :64] (B^T)
  __shared__ __attribute__((aligned(16))) __bf16 sWu2[64 * LDS_P];   // upd_w[:, 64:] (B^T)
  __shared__ __attribute__((aligned(16))) __bf16 sXb[64 * LDS_P];    // X bf16, row-major [node][feat]
  __shared__ __attribute__((aligned(16))) __bf16 sMsgT[64 * LDS_P];  // msg TRANSPOSED [feat][node] (B^T)
  __shared__ __attribute__((aligned(16))) __bf16 sPmsg[64 * LDS_P];  // parent_msg, row-major [node][feat]

  const int tid  = threadIdx.x;
  const int wave = tid >> 5;
  const int lane = tid & 31;
  const int hl   = lane >> 4;   // half-wave
  const int lr   = lane & 15;
  const int r0   = wave << 4;   // this wave's 16-row node strip
  const int mrow = r0 + hl * 8; // base row of this lane's C-layout elements

  // Stage the 4 prepacked bf16 weight matrices into padded LDS (b128 copies).
  {
    const __bf16* srcs[4] = { wsb, wsb + 4096, wsb + 2 * 4096, wsb + 3 * 4096 };
    __bf16* dsts[4] = { sAb, sWmsg, sWu1, sWu2 };
    for (int a4 = 0; a4 < 4; ++a4) {
      const __bf16* s = srcs[a4];
      __bf16* d = dsts[a4];
      for (int c = 0; c < 4; ++c) {
        int f = tid * 32 + c * 8;
        int row = f >> 6, col = f & 63;
        *(v8bf*)(d + row * LDS_P + col) = *(const v8bf*)(s + f);
      }
    }
  }

  // Load this wave's 16x64 strip of latent into registers (WMMA C layout) + bf16 LDS copy.
  const float* src = latent + (size_t)blockIdx.x * 64 * 64;
  v8f oreg[4];
  for (int nt = 0; nt < 4; ++nt) {
    int ncol = nt * 16 + lr;
    for (int r = 0; r < 8; ++r) {
      float v = src[(mrow + r) * 64 + ncol];
      oreg[nt][r] = v;
      sXb[(mrow + r) * LDS_P + ncol] = (__bf16)v;
    }
  }

  float mb[4], ub[4];
  for (int nt = 0; nt < 4; ++nt) { mb[nt] = msg_b[nt * 16 + lr]; ub[nt] = upd_b[nt * 16 + lr]; }

  __syncthreads();

  for (int it = 0; it < 2; ++it) {
    // ---- Step 1: msg = tanh(X @ msg_w^T + msg_b); store transposed -> sMsgT ----
    for (int nt = 0; nt < 4; ++nt) {
      v8f acc;
      for (int r = 0; r < 8; ++r) acc[r] = mb[nt];
      for (int kt = 0; kt < 2; ++kt) {
        v16bf a = load_a_frag(sXb, r0 + lr, kt, hl);
        v16bf b = load_b_frag(sWmsg, nt, kt, lr, hl);
        acc = WMMA_BF16(acc, a, b);
      }
      v8bf pk;
      for (int r = 0; r < 8; ++r) pk[r] = (__bf16)tanh_fast(acc[r]);
      *(v8bf*)(sMsgT + (nt * 16 + lr) * LDS_P + mrow) = pk;  // contiguous transposed store
    }
    __syncthreads();  // all rows of msg needed by every wave

    // ---- Step 2: parent_msg = Ahat @ msg (1/parent_count pre-folded into Ahat) ----
    for (int nt = 0; nt < 4; ++nt) {
      v8f acc;
      for (int r = 0; r < 8; ++r) acc[r] = 0.0f;
      for (int kt = 0; kt < 2; ++kt) {
        v16bf a = load_a_frag(sAb, r0 + lr, kt, hl);
        v16bf b = load_b_frag(sMsgT, nt, kt, lr, hl);
        acc = WMMA_BF16(acc, a, b);
      }
      for (int r = 0; r < 8; ++r)
        sPmsg[(mrow + r) * LDS_P + nt * 16 + lr] = (__bf16)acc[r];
    }
    // sPmsg rows are produced and consumed by the same wave; DS is in-order -> no barrier.

    // ---- Step 3: update = tanh([X ; pmsg] @ upd_w^T + upd_b); X = 0.7X + 0.3*update ----
    v8bf xnew[4];
    for (int nt = 0; nt < 4; ++nt) {
      v8f acc;
      for (int r = 0; r < 8; ++r) acc[r] = ub[nt];
      for (int kt = 0; kt < 2; ++kt) {
        v16bf a = load_a_frag(sXb, r0 + lr, kt, hl);
        v16bf b = load_b_frag(sWu1, nt, kt, lr, hl);
        acc = WMMA_BF16(acc, a, b);
      }
      for (int kt = 0; kt < 2; ++kt) {
        v16bf a = load_a_frag(sPmsg, r0 + lr, kt, hl);
        v16bf b = load_b_frag(sWu2, nt, kt, lr, hl);
        acc = WMMA_BF16(acc, a, b);
      }
      for (int r = 0; r < 8; ++r) {
        float o = 0.7f * oreg[nt][r] + 0.3f * tanh_fast(acc[r]);
        oreg[nt][r] = o;
        xnew[nt][r] = (__bf16)o;
      }
    }
    // Commit new X only after all N-tiles consumed the old X.
    for (int nt = 0; nt < 4; ++nt)
      for (int r = 0; r < 8; ++r)
        sXb[(mrow + r) * LDS_P + nt * 16 + lr] = xnew[nt][r];
    __syncthreads();  // sMsgT fully consumed before next iteration overwrites it
  }

  float* dst = out + (size_t)blockIdx.x * 64 * 64;
  for (int nt = 0; nt < 4; ++nt) {
    int ncol = nt * 16 + lr;
    for (int r = 0; r < 8; ++r)
      dst[(mrow + r) * 64 + ncol] = oreg[nt][r];
  }
}

extern "C" void kernel_launch(void* const* d_in, const int* in_sizes, int n_in,
                              void* d_out, int out_size, void* d_ws, size_t ws_size,
                              hipStream_t stream) {
  const float* latent = (const float*)d_in[0];
  const float* W      = (const float*)d_in[1];
  const float* msg_w  = (const float*)d_in[2];
  const float* msg_b  = (const float*)d_in[3];
  const float* upd_w  = (const float*)d_in[4];
  const float* upd_b  = (const float*)d_in[5];
  __bf16* wsb = (__bf16*)d_ws;  // 4 x 64x64 bf16 = 32 KB of scratch

  gnn_prep_kernel<<<1, 256, 0, stream>>>(W, msg_w, upd_w, wsb);
  gnn_main_kernel<<<4096, 128, 0, stream>>>(latent, msg_b, upd_b, wsb, (float*)d_out);
}
